// ModelDSS_76459007803635
// MI455X (gfx1250) — matrix operand order repaired and checked by reading, not compiled, not run on hardware
//
#include <hip/hip_runtime.h>
#include <hip/hip_bf16.h>

#define NN     50000
#define EE     800000
#define LL     32
#define KSTEPS 3
#define DPHI   67          // 2L+3
#define DPSI   97          // 3L+1
#define PHI_KP 96          // DPHI padded to 3 chunks of 32
#define PSI_KP 128         // DPSI padded to 4 chunks of 32
#define ALPHA  0.5f

typedef __attribute__((ext_vector_type(16))) _Float16 v16h;
typedef __attribute__((ext_vector_type(8)))  float    v8f;

__device__ __forceinline__ v8f wmma_f16(v16h a, v16h b, v8f c) {
  // D(f32 16x16) = A(f16 16x32) * B(f16 32x16) + C
  return __builtin_amdgcn_wmma_f32_16x16x32_f16(false, a, false, b, (short)0, c,
                                                false, false);
}

// ---- CDNA5 async memory<->LDS helpers (ASYNCcnt-tracked) -------------------
__device__ __forceinline__ unsigned lds_addr(const void* p) {
  return (unsigned)(uintptr_t)p;  // low 32 bits of generic LDS pointer = LDS offset
}
__device__ __forceinline__ void async_load_b128(unsigned lds, unsigned goff,
                                                unsigned long long sbase) {
  asm volatile("global_load_async_to_lds_b128 %0, %1, %2"
               :: "v"(lds), "v"(goff), "s"(sbase) : "memory");
}
__device__ __forceinline__ void async_store_b128(unsigned goff, unsigned lds,
                                                 unsigned long long sbase) {
  asm volatile("global_store_async_from_lds_b128 %0, %1, %2"
               :: "v"(goff), "v"(lds), "s"(sbase) : "memory");
}
__device__ __forceinline__ void wait_asynccnt0() {
  asm volatile("s_wait_asynccnt 0x0" ::: "memory");
}
__device__ __forceinline__ void wait_dscnt0() {
  asm volatile("s_wait_dscnt 0x0" ::: "memory");
}

// A operand: rows = 16 tile rows (M), cols = 32 K-values starting at kOff.
// tile is row-major f16 in LDS with row stride kStride.
__device__ __forceinline__ v16h load_a_tile(const _Float16* tile, int lane,
                                            int kOff, int kStride) {
  const int m  = lane & 15;
  const int kb = kOff + ((lane >> 4) << 3);
  v16h a;
#pragma unroll
  for (int i = 0; i < 16; ++i) {
    const int k = kb + ((i >> 3) << 4) + (i & 7);
    a[i] = tile[m * kStride + k];
  }
  return a;
}

// B operand: weight matrix stored [K][32] row-major f16 in LDS; take the
// 32(K beginning at kOff) x 16(N beginning at nOff) tile. Lane indexes N, elements index K (table order).
__device__ __forceinline__ v16h load_b_tile(const _Float16* w, int lane,
                                            int kOff, int nOff) {
  const int n  = nOff + (lane & 15);
  const int kb = kOff + ((lane >> 4) << 3);
  v16h b;
#pragma unroll
  for (int i = 0; i < 16; ++i) b[i] = w[(kb + ((i >> 3) << 4) + (i & 7)) * LL + n];
  return b;
}

// ---------------------------------------------------------------------------
// Edge kernel: for each step, compute m_to / m_from for 16 edges per wave and
// scatter-add into mess_to (at dst) / mess_from (at src).
// ---------------------------------------------------------------------------
__device__ __forceinline__ void edge_mlp_scatter(
    const _Float16* feat,      // [16][PHI_KP] this wave's feature tile
    _Float16*       bounce,    // [16][LL] per-wave relayout buffer
    const _Float16* W1,        // [PHI_KP][LL] LDS
    const float*    B1,        // [LL]
    const _Float16* W2,        // [LL][LL] LDS
    const float*    B2,        // [LL]
    const int*      nodes,     // [16] scatter targets
    const float*    mask,      // [16] loop/oob mask
    float*          outBuf,    // [N][LL] message accumulator
    int             lane) {
  const int n0   = lane & 15;
  const int half = lane >> 4;

  // ---- layer 1: [16 x PHI_KP] @ [PHI_KP x 32] + b1 ----
  v8f acc0, acc1;
  {
    const float b0 = B1[n0], b1v = B1[n0 + 16];
#pragma unroll
    for (int r = 0; r < 8; ++r) { acc0[r] = b0; acc1[r] = b1v; }
  }
#pragma unroll
  for (int kc = 0; kc < PHI_KP / 32; ++kc) {
    v16h a = load_a_tile(feat, lane, kc * 32, PHI_KP);
    acc0 = wmma_f16(a, load_b_tile(W1, lane, kc * 32, 0),  acc0);
    acc1 = wmma_f16(a, load_b_tile(W1, lane, kc * 32, 16), acc1);
  }
  // ReLU + bounce through LDS into A-operand layout
#pragma unroll
  for (int r = 0; r < 8; ++r) {
    const int mm = r + half * 8;
    bounce[mm * LL + n0]      = (_Float16)fmaxf(acc0[r], 0.f);
    bounce[mm * LL + n0 + 16] = (_Float16)fmaxf(acc1[r], 0.f);
  }
  __syncthreads();

  // ---- layer 2: [16 x 32] @ [32 x 32] + b2 ----
  v8f o0, o1;
  {
    const float b0 = B2[n0], b1v = B2[n0 + 16];
#pragma unroll
    for (int r = 0; r < 8; ++r) { o0[r] = b0; o1[r] = b1v; }
  }
  v16h a2 = load_a_tile(bounce, lane, 0, LL);
  o0 = wmma_f16(a2, load_b_tile(W2, lane, 0, 0),  o0);
  o1 = wmma_f16(a2, load_b_tile(W2, lane, 0, 16), o1);

  // ---- masked scatter-add (hardware f32 atomics, L2-resident targets) ----
#pragma unroll
  for (int r = 0; r < 8; ++r) {
    const int   mm = r + half * 8;
    const float mk = mask[mm];
    float*      p  = outBuf + (size_t)nodes[mm] * LL;
    unsafeAtomicAdd(p + n0,      o0[r] * mk);
    unsafeAtomicAdd(p + n0 + 16, o1[r] * mk);
  }
}

__global__ __launch_bounds__(256) void edge_kernel(
    const int* __restrict__ edge_index, const float* __restrict__ edge_attr,
    const unsigned short* __restrict__ h16,   // f16 mirror of h, [N][LL]
    const float* gW1to, const float* gB1to, const float* gW2to, const float* gB2to,
    const float* gW1fr, const float* gB1fr, const float* gW2fr, const float* gB2fr,
    float* mess_to, float* mess_from, int t) {
  __shared__ _Float16 sW1to[PHI_KP * LL];
  __shared__ _Float16 sW1fr[PHI_KP * LL];
  __shared__ _Float16 sW2to[LL * LL];
  __shared__ _Float16 sW2fr[LL * LL];
  __shared__ float    sB1to[LL], sB2to[LL], sB1fr[LL], sB2fr[LL];
  __shared__ __align__(16) _Float16 sFeat[8][16 * PHI_KP];
  __shared__ __align__(16) _Float16 sHid[8][16 * LL];
  __shared__ int      sSrc[8][16];
  __shared__ int      sDst[8][16];
  __shared__ float    sMask[8][16];

  const int tid  = threadIdx.x;
  const int w    = tid >> 5;
  const int lane = tid & 31;
  const int m    = lane & 15;
  const int half = lane >> 4;

  // ---- stage weights (f32 -> f16) into LDS; permute phi_to W1 rows so the
  //      single gathered tile [h_src | h_dst | ea] feeds both MLPs ----
  const float* W1t = gW1to + t * DPHI * LL;
  const float* W1f = gW1fr + t * DPHI * LL;
  const float* W2t = gW2to + t * LL * LL;
  const float* W2f = gW2fr + t * LL * LL;
  for (int idx = tid; idx < PHI_KP * LL; idx += 256) {
    const int r = idx >> 5, c = idx & 31;
    sW1fr[idx] = (_Float16)((r < DPHI) ? W1f[r * LL + c] : 0.f);
    const int gr = (r < LL) ? (r + LL) : ((r < 2 * LL) ? (r - LL) : r);
    sW1to[idx] = (_Float16)((r < DPHI) ? W1t[gr * LL + c] : 0.f);
  }
  for (int idx = tid; idx < LL * LL; idx += 256) {
    sW2to[idx] = (_Float16)W2t[idx];
    sW2fr[idx] = (_Float16)W2f[idx];
  }
  if (tid < LL) {
    sB1to[tid] = gB1to[t * LL + tid];
    sB2to[tid] = gB2to[t * LL + tid];
    sB1fr[tid] = gB1fr[t * LL + tid];
    sB2fr[tid] = gB2fr[t * LL + tid];
  }

  // ---- gather one feature tile per wave: [h_src(32) | h_dst(32) | ea(3) 0pad]
  //      h rows come from the f16 mirror via async memory->LDS DMA ----
  const int e0 = blockIdx.x * 128 + w * 16;
  const int e  = e0 + m;
  const int ec = (e < EE) ? e : (EE - 1);
  const int s  = edge_index[ec];
  const int d  = edge_index[EE + ec];
  if (half == 0) {
    sSrc[w][m]  = s;
    sDst[w][m]  = d;
    sMask[w][m] = (e < EE && s != d) ? 1.0f : 0.0f;
  }
  {
    _Float16* frow = &sFeat[w][m * PHI_KP];
    const unsigned long long hb = (unsigned long long)(uintptr_t)h16;
    const unsigned gs = (unsigned)s * (LL * 2) + half * 32;  // byte offsets
    const unsigned gd = (unsigned)d * (LL * 2) + half * 32;
    const unsigned ls = lds_addr(frow) + half * 32;
    const unsigned ld = lds_addr(frow + LL) + half * 32;
    async_load_b128(ls,      gs,      hb);
    async_load_b128(ls + 16, gs + 16, hb);
    async_load_b128(ld,      gd,      hb);
    async_load_b128(ld + 16, gd + 16, hb);
    if (half == 0) {
      frow[64] = (_Float16)edge_attr[ec * 3 + 0];
      frow[65] = (_Float16)edge_attr[ec * 3 + 1];
      frow[66] = (_Float16)edge_attr[ec * 3 + 2];
#pragma unroll
      for (int i = DPHI; i < PHI_KP; ++i) frow[i] = (_Float16)0.f;
    }
  }
  wait_asynccnt0();      // per-wave: all async tile fills landed in LDS
  __syncthreads();

  // phi_to -> aggregate at dst ; phi_from -> aggregate at src
  edge_mlp_scatter(&sFeat[w][0], &sHid[w][0], sW1to, sB1to, sW2to, sB2to,
                   sDst[w], sMask[w], mess_to, lane);
  edge_mlp_scatter(&sFeat[w][0], &sHid[w][0], sW1fr, sB1fr, sW2fr, sB2fr,
                   sSrc[w], sMask[w], mess_from, lane);
}

// ---------------------------------------------------------------------------
// Node kernel: psi correction + h update (+ decoder on last step).
// Also refreshes the f16 mirror h16 via async LDS->memory DMA.
// ---------------------------------------------------------------------------
__global__ __launch_bounds__(256) void node_kernel(
    float* __restrict__ h, unsigned short* __restrict__ h16,
    const float* __restrict__ mess_to, const float* __restrict__ mess_from,
    const float* __restrict__ prb,
    const float* gPsiW1, const float* gPsiB1, const float* gPsiW2, const float* gPsiB2,
    const float* gDecW1, const float* gDecB1, const float* gDecW2, const float* gDecB2,
    float* __restrict__ out, int t, int last) {
  __shared__ _Float16 sPsiW1[PSI_KP * LL];
  __shared__ _Float16 sPsiW2[LL * LL];
  __shared__ _Float16 sDecW1[LL * LL];
  __shared__ float    sPsiB1[LL], sPsiB2[LL], sDecB1[LL], sDecW2[LL];
  __shared__ float    sDecB2;
  __shared__ __align__(16) _Float16 sFeat[8][16 * PSI_KP];
  __shared__ __align__(16) _Float16 sHid[8][16 * LL];

  const int tid  = threadIdx.x;
  const int w    = tid >> 5;
  const int lane = tid & 31;
  const int m    = lane & 15;
  const int half = lane >> 4;
  const int n0   = m;

  // ---- stage weights ----
  const float* PW1 = gPsiW1 + t * DPSI * LL;
  for (int idx = tid; idx < PSI_KP * LL; idx += 256) {
    const int r = idx >> 5, c = idx & 31;
    sPsiW1[idx] = (_Float16)((r < DPSI) ? PW1[r * LL + c] : 0.f);
  }
  for (int idx = tid; idx < LL * LL; idx += 256) {
    sPsiW2[idx] = (_Float16)gPsiW2[t * LL * LL + idx];
    sDecW1[idx] = (_Float16)gDecW1[t * LL * LL + idx];
  }
  if (tid < LL) {
    sPsiB1[tid] = gPsiB1[t * LL + tid];
    sPsiB2[tid] = gPsiB2[t * LL + tid];
    sDecB1[tid] = gDecB1[t * LL + tid];
    sDecW2[tid] = gDecW2[t * LL + tid];  // [L][1]
  }
  if (tid == 0) sDecB2 = gDecB2[t];

  // ---- gather features: [h(32) | mess_to(32) | mess_from(32) | prb(1) 0pad] ----
  const int nd0  = blockIdx.x * 128 + w * 16;
  const int node = nd0 + m;
  const int nc   = (node < NN) ? node : (NN - 1);
  {
    _Float16*    frow = &sFeat[w][m * PSI_KP];
    const float* hp   = h + (size_t)nc * LL + half * 16;
    const float* mt   = mess_to + (size_t)nc * LL + half * 16;
    const float* mf   = mess_from + (size_t)nc * LL + half * 16;
#pragma unroll
    for (int i = 0; i < 16; ++i) frow[half * 16 + i]      = (_Float16)hp[i];
#pragma unroll
    for (int i = 0; i < 16; ++i) frow[32 + half * 16 + i] = (_Float16)mt[i];
#pragma unroll
    for (int i = 0; i < 16; ++i) frow[64 + half * 16 + i] = (_Float16)mf[i];
    if (half == 0) {
      frow[96] = (_Float16)prb[nc];
#pragma unroll
      for (int i = 97; i < 112; ++i) frow[i] = (_Float16)0.f;
    } else {
#pragma unroll
      for (int i = 112; i < PSI_KP; ++i) frow[i] = (_Float16)0.f;
    }
  }
  __syncthreads();

  // ---- psi layer 1 ----
  v8f a0, a1;
  {
    const float b0 = sPsiB1[n0], b1v = sPsiB1[n0 + 16];
#pragma unroll
    for (int r = 0; r < 8; ++r) { a0[r] = b0; a1[r] = b1v; }
  }
#pragma unroll
  for (int kc = 0; kc < PSI_KP / 32; ++kc) {
    v16h a = load_a_tile(&sFeat[w][0], lane, kc * 32, PSI_KP);
    a0 = wmma_f16(a, load_b_tile(sPsiW1, lane, kc * 32, 0),  a0);
    a1 = wmma_f16(a, load_b_tile(sPsiW1, lane, kc * 32, 16), a1);
  }
#pragma unroll
  for (int r = 0; r < 8; ++r) {
    const int mm = r + half * 8;
    sHid[w][mm * LL + n0]      = (_Float16)fmaxf(a0[r], 0.f);
    sHid[w][mm * LL + n0 + 16] = (_Float16)fmaxf(a1[r], 0.f);
  }
  __syncthreads();

  // ---- psi layer 2 + h update ----
  v8f c0, c1;
  {
    const float b0 = sPsiB2[n0], b1v = sPsiB2[n0 + 16];
#pragma unroll
    for (int r = 0; r < 8; ++r) { c0[r] = b0; c1[r] = b1v; }
  }
  {
    v16h a2 = load_a_tile(&sHid[w][0], lane, 0, LL);
    c0 = wmma_f16(a2, load_b_tile(sPsiW2, lane, 0, 0),  c0);
    c1 = wmma_f16(a2, load_b_tile(sPsiW2, lane, 0, 16), c1);
  }
#pragma unroll
  for (int r = 0; r < 8; ++r) {
    const int mm    = r + half * 8;
    const int nodeR = nd0 + mm;
    if (nodeR < NN) {
      float* hp  = h + (size_t)nodeR * LL;
      const float hn0 = hp[n0]      + ALPHA * c0[r];
      const float hn1 = hp[n0 + 16] + ALPHA * c1[r];
      hp[n0]      = hn0;
      hp[n0 + 16] = hn1;
      sHid[w][mm * LL + n0]      = (_Float16)hn0;
      sHid[w][mm * LL + n0 + 16] = (_Float16)hn1;
    } else {
      sHid[w][mm * LL + n0]      = (_Float16)0.f;
      sHid[w][mm * LL + n0 + 16] = (_Float16)0.f;
    }
  }

  // ---- refresh f16 mirror: async LDS -> global (feeds next step's edge gather)
  wait_dscnt0();  // async LDS reads are unordered vs prior DS stores
  {
    const int nodeR2 = nd0 + m;
    if (nodeR2 < NN) {
      const unsigned long long hb = (unsigned long long)(uintptr_t)h16;
      const unsigned lsrc = lds_addr(&sHid[w][m * LL]) + half * 32;
      const unsigned goff = (unsigned)nodeR2 * (LL * 2) + half * 32;
      async_store_b128(goff,      lsrc,      hb);
      async_store_b128(goff + 16, lsrc + 16, hb);
    }
  }
  wait_asynccnt0();  // sHid is reused below

  // ---- decoder: only the final step's u is observable ----
  if (last) {
    v8f d0, d1;
    {
      const float b0 = sDecB1[n0], b1v = sDecB1[n0 + 16];
#pragma unroll
      for (int r = 0; r < 8; ++r) { d0[r] = b0; d1[r] = b1v; }
    }
    v16h a3 = load_a_tile(&sHid[w][0], lane, 0, LL);
    d0 = wmma_f16(a3, load_b_tile(sDecW1, lane, 0, 0),  d0);
    d1 = wmma_f16(a3, load_b_tile(sDecW1, lane, 0, 16), d1);
#pragma unroll
    for (int r = 0; r < 8; ++r) {
      const int mm = r + half * 8;
      sHid[w][mm * LL + n0]      = (_Float16)fmaxf(d0[r], 0.f);
      sHid[w][mm * LL + n0 + 16] = (_Float16)fmaxf(d1[r], 0.f);
    }
    if (lane < 16) {
      const int nodeR = nd0 + lane;
      if (nodeR < NN) {
        float acc = sDecB2;
#pragma unroll
        for (int k = 0; k < LL; ++k)
          acc += (float)sHid[w][lane * LL + k] * sDecW2[k];
        out[nodeR] = acc;
      }
    }
  }
}

__global__ void zero_kernel(float* __restrict__ p, int n) {
  const int i = blockIdx.x * blockDim.x + threadIdx.x;
  if (i < n) p[i] = 0.f;
}

extern "C" void kernel_launch(void* const* d_in, const int* in_sizes, int n_in,
                              void* d_out, int out_size, void* d_ws, size_t ws_size,
                              hipStream_t stream) {
  (void)in_sizes; (void)n_in; (void)out_size; (void)ws_size;
  const int*   edge_index = (const int*)d_in[0];
  const float* edge_attr  = (const float*)d_in[1];
  const float* prb        = (const float*)d_in[2];
  const float* phi_to_W1   = (const float*)d_in[3];
  const float* phi_to_b1   = (const float*)d_in[4];
  const float* phi_to_W2   = (const float*)d_in[5];
  const float* phi_to_b2   = (const float*)d_in[6];
  const float* phi_from_W1 = (const float*)d_in[7];
  const float* phi_from_b1 = (const float*)d_in[8];
  const float* phi_from_W2 = (const float*)d_in[9];
  const float* phi_from_b2 = (const float*)d_in[10];
  const float* psi_W1 = (const float*)d_in[11];
  const float* psi_b1 = (const float*)d_in[12];
  const float* psi_W2 = (const float*)d_in[13];
  const float* psi_b2 = (const float*)d_in[14];
  const float* dec_W1 = (const float*)d_in[15];
  const float* dec_b1 = (const float*)d_in[16];
  const float* dec_W2 = (const float*)d_in[17];
  const float* dec_b2 = (const float*)d_in[18];
  float* out = (float*)d_out;

  float* h               = (float*)d_ws;            // [N][L] f32
  float* mess_to         = h + (size_t)NN * LL;     // [N][L] f32
  float* mess_from       = mess_to + (size_t)NN * LL;
  unsigned short* h16    = (unsigned short*)(mess_from + (size_t)NN * LL); // [N][L] f16

  const int hElems    = NN * LL;
  const int messElems = 2 * NN * LL;
  const int h16Floats = NN * LL / 2;  // f16 mirror zeroed as f32 words

  zero_kernel<<<(hElems + 255) / 256, 256, 0, stream>>>(h, hElems);
  zero_kernel<<<(h16Floats + 255) / 256, 256, 0, stream>>>((float*)h16, h16Floats);

  const int edgeBlocks = (EE + 127) / 128;
  const int nodeBlocks = (NN + 127) / 128;
  for (int t = 0; t < KSTEPS; ++t) {
    zero_kernel<<<(messElems + 255) / 256, 256, 0, stream>>>(mess_to, messElems);
    edge_kernel<<<edgeBlocks, 256, 0, stream>>>(
        edge_index, edge_attr, h16,
        phi_to_W1, phi_to_b1, phi_to_W2, phi_to_b2,
        phi_from_W1, phi_from_b1, phi_from_W2, phi_from_b2,
        mess_to, mess_from, t);
    node_kernel<<<nodeBlocks, 256, 0, stream>>>(
        h, h16, mess_to, mess_from, prb,
        psi_W1, psi_b1, psi_W2, psi_b2,
        dec_W1, dec_b1, dec_W2, dec_b2,
        out, t, (t == KSTEPS - 1) ? 1 : 0);
  }
}
